// Decoder_90623809945636
// MI455X (gfx1250) — matrix operand order repaired and checked by reading, not compiled, not run on hardware
//
#include <hip/hip_runtime.h>
#include <hip/hip_bf16.h>

// ---------------------------------------------------------------------------
// Problem constants (reference): VOCAB=32000 EMB=512 H=1024 B=64 S=128
// ---------------------------------------------------------------------------
#define VOCAB 32000
#define EMB   512
#define HDIM  1024
#define BATCH 64
#define SEQ   128

typedef __attribute__((ext_vector_type(16))) __bf16 bf16x16;
typedef __attribute__((ext_vector_type(8)))  float  f32x8;
typedef __attribute__((ext_vector_type(4)))  unsigned int u32x4;
typedef __attribute__((ext_vector_type(4)))  int    i32x4;
typedef __attribute__((ext_vector_type(8)))  int    i32x8;
typedef unsigned short ushort_t;

#if __has_builtin(__builtin_amdgcn_tensor_load_to_lds)
#define HAVE_TDM 1
#else
#define HAVE_TDM 0
#endif

// fp32 pair -> packed bf16 dword (truncation; WMMA accumulates in f32)
__device__ __forceinline__ unsigned pk2(float a, float b) {
    return (__builtin_bit_cast(unsigned, a) >> 16) |
           (__builtin_bit_cast(unsigned, b) & 0xffff0000u);
}

__device__ __forceinline__ f32x8 wmma_bf16(bf16x16 a, bf16x16 b, f32x8 c) {
    return __builtin_amdgcn_wmma_f32_16x16x32_bf16(
        /*neg_a=*/false, a, /*neg_b=*/false, b,
        /*c_mod=*/(short)0, c, /*reuse_a=*/false, /*reuse_b=*/false);
}

// --- TDM: DMA a 1-D run of fp32 from global into LDS (wave-uniform args) ---
// D# per CDNA5 ISA 8.3/8.4: group0 = {count/type | lds_addr | global_addr},
// group1 = {data_size, tensor_dim0/1, tile_dim0/1, stride0}. 2-D descriptor:
// groups 2/3 zero. Tracked with TENSORcnt.
__device__ __forceinline__ void tdm_load_f32_row(const float* gsrc,
                                                 unsigned ldsByteOff,
                                                 unsigned nElems) {
#if HAVE_TDM
    unsigned long long ga = (unsigned long long)gsrc;
    u32x4 g0;
    g0[0] = 1u;                                       // count=1, user mode
    g0[1] = ldsByteOff;                               // lds_addr (bytes)
    g0[2] = (unsigned)(ga & 0xffffffffu);             // global_addr[31:0]
    g0[3] = (unsigned)((ga >> 32) & 0x01ffffffu)      // global_addr[56:32]
          | (2u << 30);                               // type=2 ("image")
    i32x8 g1;
    g1[0] = (int)(2u << 16);                          // wg_mask=0, data_size=4B
    g1[1] = (int)((nElems & 0xffffu) << 16);          // tensor_dim0[15:0]
    g1[2] = (int)((nElems >> 16) & 0xffffu) | (1 << 16); // dim0 hi | tensor_dim1=1
    g1[3] = (int)((nElems & 0xffffu) << 16);          // tile_dim0 = nElems
    g1[4] = 1;                                        // tile_dim1 = 1
    g1[5] = (int)nElems;                              // tensor_dim0_stride
    g1[6] = 0;
    g1[7] = 0;
    i32x4 z4 = {0, 0, 0, 0};
#if defined(__clang_major__) && __clang_major__ >= 23
    i32x8 z8 = {0, 0, 0, 0, 0, 0, 0, 0};
    __builtin_amdgcn_tensor_load_to_lds(g0, g1, z4, z4, z8, 0);
#else
    __builtin_amdgcn_tensor_load_to_lds(g0, g1, z4, z4, 0);
#endif
#if __has_builtin(__builtin_amdgcn_s_wait_tensorcnt)
    __builtin_amdgcn_s_wait_tensorcnt((short)0);      // s_wait_tensorcnt 0
#endif
#endif
}

// --- B-fragment from global fp32 (row-major, ldb): 32x16 bf16 --------------
// Lane L: n = L%16, Kbase = 8*(L/16). Element i -> K = Kbase + (i&7) + 16*(i>>3).
__device__ __forceinline__ bf16x16 load_b_frag(const float* __restrict__ B,
                                               int ldb, int k0, int n0, int lane) {
    int n  = lane & 15;
    int kb = (lane >> 4) << 3;
    const float* col = B + (size_t)(k0 + kb) * ldb + n0 + n;
    union { bf16x16 v; unsigned u[8]; } u;
#pragma unroll
    for (int i = 0; i < 8; ++i)
        u.u[i] = pk2(col[(size_t)(2 * i + ((i >> 2) << 3)) * ldb], col[(size_t)(2 * i + 1 + ((i >> 2) << 3)) * ldb]);
    return u.v;
}

// --- Stage one A-fragment (16x32 bf16) into fragment-ready LDS -------------
// A-fragment ISA layout: lane L (half=L/16, m=L%16), element i ->
//   K = (i&7) + 8*half + 16*(i>>3)  => two 8-float contiguous runs per lane.
// LDS layout: fragment f, lane L -> 32 contiguous bytes at (f*32+L)*32.
__device__ __forceinline__ void stage_a_frag(const float* __restrict__ rowBase,
                                             int k0, int half,
                                             ushort_t* __restrict__ fragLds,
                                             int slot /* = f*32+lane */) {
    const float* r = rowBase + k0 + (half << 3);
    float4 f0 = *(const float4*)(r);
    float4 f1 = *(const float4*)(r + 4);
    float4 f2 = *(const float4*)(r + 16);
    float4 f3 = *(const float4*)(r + 20);
    uint4 q0 = { pk2(f0.x, f0.y), pk2(f0.z, f0.w), pk2(f1.x, f1.y), pk2(f1.z, f1.w) };
    uint4 q1 = { pk2(f2.x, f2.y), pk2(f2.z, f2.w), pk2(f3.x, f3.y), pk2(f3.z, f3.w) };
    uint4* dst = (uint4*)(fragLds + (size_t)slot * 16);
    dst[0] = q0;                      // ds_store_b128
    dst[1] = q1;                      // ds_store_b128
}

// --- Read one A-fragment back from LDS: 2x ds_load_b128 --------------------
__device__ __forceinline__ bf16x16 lds_frag(const ushort_t* __restrict__ fragLds,
                                            int f, int lane) {
    const uint4* p = (const uint4*)(fragLds + (size_t)((f << 5) + lane) * 16);
    union { bf16x16 v; uint4 q[2]; } u;
    u.q[0] = p[0];
    u.q[1] = p[1];
    return u.v;
}

// ---------------------------------------------------------------------------
// Generic GEMM: C(64 x N) = A(64 x K) @ B(K x N) + bias.
// Block = 8 waves. A is staged bf16 fragment-ready into LDS once per block
// (4 M-tiles x K/32 fragments); each wave owns one 16-wide N-tile and streams
// B from global exactly once (critical for out_k = 131 MB).
// Dynamic LDS: K*128 bytes. K%32==0, N%16==0.
// ---------------------------------------------------------------------------
__global__ void gemm64_bf16_wmma(const float* __restrict__ A,
                                 const float* __restrict__ B,
                                 const float* __restrict__ bias,
                                 float* __restrict__ C,
                                 int K, int N) {
    extern __shared__ char smem[];
    ushort_t* fragA = (ushort_t*)smem;           // [4*KSTEPS*32][16]
    const int KSTEPS = K >> 5;
    const int lane   = threadIdx.x & 31;

    // ---- cooperative fragment-ready staging of A (all waves) ----
    for (int t = threadIdx.x; t < 4 * KSTEPS * 32; t += blockDim.x) {
        int ln    = t & 31;
        int idx   = t >> 5;
        int kstep = idx % KSTEPS;
        int mtile = idx / KSTEPS;
        int m     = (mtile << 4) + (ln & 15);
        stage_a_frag(A + (size_t)m * K, kstep << 5, ln >> 4, fragA, t);
    }
    __syncthreads();

    const int tile   = blockIdx.x * (blockDim.x >> 5) + (threadIdx.x >> 5);
    const int nTiles = N >> 4;
    if (tile >= nTiles) return;                  // wave-uniform
    const int n0 = tile << 4;

    f32x8 c0 = {}, c1 = {}, c2 = {}, c3 = {};
    for (int ks = 0; ks < KSTEPS; ++ks) {
        if (ks + 1 < KSTEPS)                     // global_prefetch_b8 next B slab
            __builtin_prefetch(B + (size_t)((ks + 1) << 5) * N + n0 + (lane & 15), 0, 1);
        bf16x16 bf = load_b_frag(B, N, ks << 5, n0, lane);
        c0 = wmma_bf16(lds_frag(fragA, 0 * KSTEPS + ks, lane), bf, c0);
        c1 = wmma_bf16(lds_frag(fragA, 1 * KSTEPS + ks, lane), bf, c1);
        c2 = wmma_bf16(lds_frag(fragA, 2 * KSTEPS + ks, lane), bf, c2);
        c3 = wmma_bf16(lds_frag(fragA, 3 * KSTEPS + ks, lane), bf, c3);
    }

    const int n    = n0 + (lane & 15);
    const int half = lane >> 4;
    const float bv = bias ? bias[n] : 0.0f;
#pragma unroll
    for (int r = 0; r < 8; ++r) {
        int m = r + 8 * half;
        C[(size_t)(m +  0) * N + n] = c0[r] + bv;
        C[(size_t)(m + 16) * N + n] = c1[r] + bv;
        C[(size_t)(m + 32) * N + n] = c2[r] + bv;
        C[(size_t)(m + 48) * N + n] = c3[r] + bv;
    }
}

// ---------------------------------------------------------------------------
// Fused attention score:
//   score[b,s] = sum_k tanh((enc[b,s]@W1)[k] + W1_b[k] + hW2[b,k]) * V_k[k] + V_b
// One 8-wave block per batch row. enc[b] (128x1024) is staged ONCE into LDS
// as 256 KB of fragment-ready bf16; hW2[b,:] (4 KB) arrives via the Tensor
// Data Mover (tensor_load_to_lds + s_wait_tensorcnt). Each wave owns 8
// N-tiles; every W1 B-fragment is amortized over 8 WMMAs (all S-tiles).
// Partial scores: shfl_xor lane reduce + ds_add_f32 across waves.
// Dynamic LDS: 256 KB + 512 B + 4 KB.
// ---------------------------------------------------------------------------
#define FRAG_BYTES ((size_t)8 * 32 * 32 * 32)          // 256 KB
#define SCORE_OFF  FRAG_BYTES                           // 512 B
#define HW2_OFF    (FRAG_BYTES + 512)                   // 4 KB
#define ATTN_LDS   (HW2_OFF + HDIM * 4)

__global__ void attn_score_kernel(const float* __restrict__ enc,
                                  const float* __restrict__ W1,
                                  const float* __restrict__ W1b,
                                  const float* __restrict__ hW2,
                                  const float* __restrict__ Vk,
                                  const float* __restrict__ Vb,
                                  float* __restrict__ score) {
    extern __shared__ char smem[];
    const int KSTEPS = HDIM >> 5;                        // 32
    ushort_t* fragA    = (ushort_t*)smem;                // 8*32*32 frags = 256 KB
    float*    score_sm = (float*)(smem + SCORE_OFF);
    float*    hW2_sm   = (float*)(smem + HW2_OFF);

    const int b    = blockIdx.x;
    const int lane = threadIdx.x & 31;
    const int w    = threadIdx.x >> 5;

    // ---- TDM: pull hW2[b,:] into LDS while VALU does the fragment staging --
#if HAVE_TDM
    if (w == 0)
        tdm_load_f32_row(hW2 + (size_t)b * HDIM, HW2_OFF, HDIM);
#else
    for (int t = threadIdx.x; t < HDIM; t += blockDim.x)
        hW2_sm[t] = hW2[(size_t)b * HDIM + t];
#endif

    // ---- stage enc[b] fragment-ready + zero the score accumulator ----
    for (int t = threadIdx.x; t < SEQ; t += blockDim.x) score_sm[t] = 0.0f;
    for (int t = threadIdx.x; t < 8 * KSTEPS * 32; t += blockDim.x) {
        int ln    = t & 31;
        int idx   = t >> 5;
        int kstep = idx % KSTEPS;
        int stile = idx / KSTEPS;
        int s     = (stile << 4) + (ln & 15);
        stage_a_frag(enc + ((size_t)b * SEQ + s) * HDIM, kstep << 5, ln >> 4, fragA, t);
    }
    __syncthreads();

    float part[8][8];
#pragma unroll
    for (int st = 0; st < 8; ++st)
#pragma unroll
        for (int r = 0; r < 8; ++r) part[st][r] = 0.0f;

    const int n    = lane & 15;
    const int half = lane >> 4;

    for (int j = 0; j < 8; ++j) {                         // 8 N-tiles per wave
        const int n0 = ((w << 3) + j) << 4;
        f32x8 c[8];
#pragma unroll
        for (int st = 0; st < 8; ++st) c[st] = (f32x8){};

        for (int ks = 0; ks < KSTEPS; ++ks) {
            bf16x16 bb = load_b_frag(W1, HDIM, ks << 5, n0, lane);
#pragma unroll
            for (int st = 0; st < 8; ++st)
                c[st] = wmma_bf16(lds_frag(fragA, st * KSTEPS + ks, lane), bb, c[st]);
        }
        const int   nn = n0 + n;
        const float bs = W1b[nn] + hW2_sm[nn];
        const float v  = Vk[nn];
#pragma unroll
        for (int st = 0; st < 8; ++st)
#pragma unroll
            for (int r = 0; r < 8; ++r)
                part[st][r] += tanhf(c[st][r] + bs) * v;
    }

    // reduce the 16 lanes sharing each s-row, then combine waves via LDS atomics
#pragma unroll
    for (int st = 0; st < 8; ++st)
#pragma unroll
        for (int r = 0; r < 8; ++r) {
            float p = part[st][r];
            p += __shfl_xor(p, 1, 32);
            p += __shfl_xor(p, 2, 32);
            p += __shfl_xor(p, 4, 32);
            p += __shfl_xor(p, 8, 32);
            if (n == 0)
                atomicAdd(&score_sm[(st << 4) + r + 8 * half], p);   // ds_add_f32
        }
    __syncthreads();
    if (threadIdx.x < SEQ)
        score[(size_t)b * SEQ + threadIdx.x] = score_sm[threadIdx.x] + Vb[0];
}

// ---------------------------------------------------------------------------
// Softmax over S, context = sum_s w*enc, attention weights to d_out,
// x = [context | emb[token]]. One 128-thread block per batch row.
// ---------------------------------------------------------------------------
__global__ void softmax_context_kernel(const float* __restrict__ score,
                                       const float* __restrict__ enc,
                                       const float* __restrict__ emb,
                                       const int* __restrict__ dec_input,
                                       float* __restrict__ attn_out,
                                       float* __restrict__ x) {
    __shared__ float red[SEQ];
    __shared__ float w[SEQ];
    const int b = blockIdx.x, t = threadIdx.x;

    const float sc = score[(size_t)b * SEQ + t];
    red[t] = sc;
    __syncthreads();
    for (int off = SEQ / 2; off > 0; off >>= 1) {
        if (t < off) red[t] = fmaxf(red[t], red[t + off]);
        __syncthreads();
    }
    const float mx = red[0];
    __syncthreads();

    const float e = __expf(sc - mx);
    red[t] = e;
    __syncthreads();
    for (int off = SEQ / 2; off > 0; off >>= 1) {
        if (t < off) red[t] += red[t + off];
        __syncthreads();
    }
    const float wgt = e / red[0];
    w[t] = wgt;
    attn_out[(size_t)b * SEQ + t] = wgt;
    __syncthreads();

    const float* encb = enc + (size_t)b * SEQ * HDIM;
    for (int h = t; h < HDIM; h += SEQ) {
        float acc = 0.0f;
#pragma unroll 4
        for (int s = 0; s < SEQ; ++s)
            acc = fmaf(w[s], encb[(size_t)s * HDIM + h], acc);
        x[(size_t)b * (HDIM + EMB) + h] = acc;
    }
    const int tok = dec_input[b];
    for (int j = t; j < EMB; j += SEQ)
        x[(size_t)b * (HDIM + EMB) + HDIM + j] = emb[(size_t)tok * EMB + j];
}

// ---------------------------------------------------------------------------
// GRU gates: state = z*h + (1-z)*tanh(xh + r*hh)
// ---------------------------------------------------------------------------
__global__ void gru_gate_kernel(const float* __restrict__ xm,
                                const float* __restrict__ hm,
                                const float* __restrict__ hidden,
                                float* __restrict__ state_ws,
                                float* __restrict__ state_out) {
    const int idx = blockIdx.x * blockDim.x + threadIdx.x;
    if (idx >= BATCH * HDIM) return;
    const int b = idx >> 10;
    const int k = idx & (HDIM - 1);
    const float* xb = xm + (size_t)b * 3 * HDIM;
    const float* hb = hm + (size_t)b * 3 * HDIM;
    const float z  = 1.0f / (1.0f + __expf(-(xb[k]        + hb[k])));
    const float r  = 1.0f / (1.0f + __expf(-(xb[HDIM + k] + hb[HDIM + k])));
    const float hc = tanhf(xb[2 * HDIM + k] + r * hb[2 * HDIM + k]);
    const float st = z * hidden[idx] + (1.0f - z) * hc;
    state_ws[idx]  = st;
    state_out[idx] = st;
}

// ---------------------------------------------------------------------------
extern "C" void kernel_launch(void* const* d_in, const int* in_sizes, int n_in,
                              void* d_out, int out_size, void* d_ws, size_t ws_size,
                              hipStream_t stream) {
    const int*   dec_input = (const int*)  d_in[0];
    const float* hidden    = (const float*)d_in[1];
    const float* enc       = (const float*)d_in[2];
    const float* emb       = (const float*)d_in[3];
    const float* W1_k      = (const float*)d_in[4];
    const float* W1_b      = (const float*)d_in[5];
    const float* W2_k      = (const float*)d_in[6];
    const float* W2_b      = (const float*)d_in[7];
    const float* V_k       = (const float*)d_in[8];
    const float* V_b       = (const float*)d_in[9];
    const float* gru_k     = (const float*)d_in[10];
    const float* gru_rk    = (const float*)d_in[11];
    const float* gru_b     = (const float*)d_in[12];
    const float* out_k     = (const float*)d_in[13];
    const float* out_b     = (const float*)d_in[14];

    // d_out layout: probs (64*32000) | state (64*1024) | attn (64*128)
    float* probs_out = (float*)d_out;
    float* state_out = probs_out + (size_t)BATCH * VOCAB;
    float* attn_out  = state_out + (size_t)BATCH * HDIM;

    // Workspace layout (floats): ~2.41 MB total
    float* ws       = (float*)d_ws;
    float* hW2      = ws;                              // 64*1024
    float* score    = hW2   + BATCH * HDIM;            // 64*128
    float* x        = score + BATCH * SEQ;             // 64*1536
    float* xm       = x     + BATCH * (HDIM + EMB);    // 64*3072
    float* hm       = xm    + BATCH * 3 * HDIM;        // 64*3072
    float* state_ws = hm    + BATCH * 3 * HDIM;        // 64*1024

    const int WPB = 8;                                 // waves per 256-thread block

    // 1) hW2 = hidden @ W2_k + W2_b            (K=1024, N=1024; LDS 128 KB)
    gemm64_bf16_wmma<<<(HDIM / 16) / WPB, 256, (size_t)HDIM * 128, stream>>>(
        hidden, W2_k, W2_b, hW2, HDIM, HDIM);

    // 2) fused attention scores (17-GFLOP GEMM; LDS ~260.5 KB)
    attn_score_kernel<<<BATCH, 256, ATTN_LDS, stream>>>(
        enc, W1_k, W1_b, hW2, V_k, V_b, score);

    // 3) softmax + context + embedding gather -> attn weights + x
    softmax_context_kernel<<<BATCH, SEQ, 0, stream>>>(
        score, enc, emb, dec_input, attn_out, x);

    // 4) xm = x @ gru_k + gru_b[0]             (K=1536, N=3072; LDS 192 KB)
    gemm64_bf16_wmma<<<(3 * HDIM / 16) / WPB, 256, (size_t)(HDIM + EMB) * 128, stream>>>(
        x, gru_k, gru_b, xm, HDIM + EMB, 3 * HDIM);

    // 5) hm = hidden @ gru_rk + gru_b[1]       (K=1024, N=3072; LDS 128 KB)
    gemm64_bf16_wmma<<<(3 * HDIM / 16) / WPB, 256, (size_t)HDIM * 128, stream>>>(
        hidden, gru_rk, gru_b + 3 * HDIM, hm, HDIM, 3 * HDIM);

    // 6) GRU gates -> state (ws copy + output copy)
    gru_gate_kernel<<<(BATCH * HDIM + 255) / 256, 256, 0, stream>>>(
        xm, hm, hidden, state_ws, state_out);

    // 7) probs = state @ out_k + out_b         (K=1024, N=32000; HBM-bound)
    gemm64_bf16_wmma<<<(VOCAB / 16) / WPB, 256, (size_t)HDIM * 128, stream>>>(
        state_ws, out_k, out_b, probs_out, HDIM, VOCAB);
}
